// DebiasedRNN_18545668784389
// MI455X (gfx1250) — matrix operand order repaired and checked by reading, not compiled, not run on hardware
//
#include <hip/hip_runtime.h>

// Problem sizes (fixed by the reference)
#define TT 512
#define BB 256
#define DD 128
#define HH 128

typedef __attribute__((ext_vector_type(16))) _Float16 v16h;
typedef __attribute__((ext_vector_type(8)))  float    v8f;

// ---------------------------------------------------------------------------
// WMMA helper: D = A(16x32 f16) * B(32x16 f16) + C(16x16 f32)
// ---------------------------------------------------------------------------
__device__ __forceinline__ v8f wmma_f16(v16h a, v16h b, v8f c) {
  return __builtin_amdgcn_wmma_f32_16x16x32_f16(
      /*neg_a=*/false, a, /*neg_b=*/false, b,
      /*c_mod=*/(short)0, c, /*reuse_a=*/false, /*reuse_b=*/false);
}

// A-operand (16x32, f16) from a row-major [*, 128] f16 buffer.
// Lane L<16: row M=L, K = kbase+{0..7, 16..23}; lanes >=16: K = kbase+{8..15, 24..31}.
__device__ __forceinline__ v16h load_a_tile(const _Float16* buf, int row, int kbase, int lane) {
  const _Float16* p = buf + row * 128 + kbase + ((lane & 16) ? 8 : 0);
  union { uint4 q[2]; v16h v; } u;
  u.q[0] = *(const uint4*)(p);        // 8 halves: K = k0 .. k0+7
  u.q[1] = *(const uint4*)(p + 16);   // 8 halves: K = k0+16 .. k0+23
  return u.v;
}

// B-operand (32x16, f16) from an N-major transposed weight buffer wt[n][k] (k contiguous).
// Lane L<16: column N = col, K = kbase+{0..7, 16..23}; lanes >=16: K = kbase+{8..15, 24..31}.
__device__ __forceinline__ v16h load_b_tile(const _Float16* wt, int col, int kbase, int lane) {
  const _Float16* p = wt + col * 128 + kbase + ((lane & 16) ? 8 : 0);
  union { uint4 q[2]; v16h v; } u;
  u.q[0] = *(const uint4*)(p);        // K = k0 .. k0+7
  u.q[1] = *(const uint4*)(p + 16);   // K = k0+16 .. k0+23
  return u.v;
}

__device__ __forceinline__ float sigmoidf_fast(float x) {
  return 1.0f / (1.0f + __expf(-x));
}

// Fragment-order layout for the precomputed gate pre-activations:
// element v of lane L of tile (t, mg, n) lives at frag_base(t,mg,n,L) + v.
// Each lane owns 8 contiguous floats (32 B) -> two b128 per access.
__device__ __forceinline__ size_t frag_base(int t, int mg, int n, int lane) {
  return ((((size_t)t * 128) + mg * 8 + n) * 32 + lane) * 8;
}

union Frag8 { float4 q[2]; v8f v; };

// ---------------------------------------------------------------------------
// Kernel A: parallel precompute of the input-half of all three gates:
//   xg[frag(t, mg, n)] = inputs[t, b, :] @ W_g[:D, :] + b_g   (g = r, u, h)
// One block = 128 rows x 128 cols x 3 gates, 256 threads (8 waves).
// x tile is read from HBM once and shared by all three gate GEMMs.
// ---------------------------------------------------------------------------
__global__ __launch_bounds__(256, 1)
void rnn_xpart_kernel(const float* __restrict__ x,
                      const float* __restrict__ Wr, const float* __restrict__ Wu,
                      const float* __restrict__ Wh,
                      const float* __restrict__ br, const float* __restrict__ bu,
                      const float* __restrict__ bh,
                      float* __restrict__ xr, float* __restrict__ xu,
                      float* __restrict__ xh) {
  extern __shared__ _Float16 smemA[];
  _Float16* xs  = smemA;               // [128 rows][128 k]
  _Float16* wtr = xs  + 128 * 128;     // [128 n][128 k] transposed, input half
  _Float16* wtu = wtr + 128 * 128;
  _Float16* wth = wtu + 128 * 128;

  const int tid = threadIdx.x;
  const size_t row0 = (size_t)blockIdx.x * 128;   // flat row in [0, T*B)
  const int t       = (int)(row0 >> 8);           // timestep
  const int mg_base = ((int)(row0 & 255)) >> 4;   // 0 or 8: M-tile base in batch

  for (int i = tid; i < 128 * 128; i += 256) {
    int r = i >> 7, k = i & 127;
    xs[i]  = (_Float16)x[(row0 + r) * DD + k];
    wtr[i] = (_Float16)Wr[k * HH + r];            // wt[n][k] = W[k][n], k < D
    wtu[i] = (_Float16)Wu[k * HH + r];
    wth[i] = (_Float16)Wh[k * HH + r];
  }
  __syncthreads();

  const int lane = tid & 31;
  const int wave = tid >> 5;               // 8 waves -> N-tile = wave
  const int col  = wave * 16 + (lane & 15);
  const float b0r = br[col], b0u = bu[col], b0h = bh[col];

  for (int m = 0; m < 8; ++m) {            // 8 M-subtiles of 16 rows
    v8f ar, au, ah;
    for (int v = 0; v < 8; ++v) { ar[v] = b0r; au[v] = b0u; ah[v] = b0h; }
    for (int kk = 0; kk < 4; ++kk) {
      v16h a = load_a_tile(xs, 16 * m + (lane & 15), 32 * kk, lane);  // shared A
      ar = wmma_f16(a, load_b_tile(wtr, col, 32 * kk, lane), ar);
      au = wmma_f16(a, load_b_tile(wtu, col, 32 * kk, lane), au);
      ah = wmma_f16(a, load_b_tile(wth, col, 32 * kk, lane), ah);
    }
    const size_t fb = frag_base(t, mg_base + m, wave, lane);
    Frag8 fr, fu, fh;
    fr.v = ar; fu.v = au; fh.v = ah;
    *(float4*)(xr + fb)     = fr.q[0];  *(float4*)(xr + fb + 4) = fr.q[1];
    *(float4*)(xu + fb)     = fu.q[0];  *(float4*)(xu + fb + 4) = fu.q[1];
    *(float4*)(xh + fb)     = fh.q[0];  *(float4*)(xh + fb + 4) = fh.q[1];
  }
}

// ---------------------------------------------------------------------------
// Kernel B: persistent single-workgroup recurrence.
// 1024 threads = 32 waves. Wave w owns M-tile m = w>>1 and the 4 N-tiles
// nbase..nbase+3 (nbase = (w&1)*4) of the [256 x 128] hidden state, kept in
// fp32 registers across all 512 steps. One LDS A-load per kk feeds 8 WMMAs
// (phase 1) / 4 WMMAs (phase 2).
// ---------------------------------------------------------------------------
__global__ __launch_bounds__(1024, 1)
void rnn_recurrent_kernel(const float* __restrict__ att,
                          const int*   __restrict__ lengths,
                          const float* __restrict__ xr,
                          const float* __restrict__ xu,
                          const float* __restrict__ xh,
                          const float* __restrict__ Wr,
                          const float* __restrict__ Wu,
                          const float* __restrict__ Wh,
                          float* __restrict__ out) {
  extern __shared__ _Float16 smem[];
  _Float16* h16  = smem;                   // [256][128] fp16 copy of h (A operand)
  _Float16* rh16 = h16  + 256 * 128;       // [256][128] r*h (A operand for h_hat)
  _Float16* wtr  = rh16 + 256 * 128;       // [128 n][128 k] recurrent half, transposed
  _Float16* wtu  = wtr  + 128 * 128;
  _Float16* wth  = wtu  + 128 * 128;
  int*      lens = (int*)(wth + 128 * 128);  // [256]
  float*    attb = (float*)(lens + 256);     // [2][256] double-buffered att row

  const int tid = threadIdx.x;

  // Load recurrent weight halves (rows D..D+H-1), transposed, fp16.
  for (int i = tid; i < 128 * 128; i += 1024) {
    int n = i >> 7, k = i & 127;
    wtr[n * 128 + k] = (_Float16)Wr[(DD + k) * HH + n];
    wtu[n * 128 + k] = (_Float16)Wu[(DD + k) * HH + n];
    wth[n * 128 + k] = (_Float16)Wh[(DD + k) * HH + n];
  }
  for (int i = tid; i < 256 * 128; i += 1024) h16[i] = (_Float16)0.0f;
  if (tid < BB) {
    lens[tid] = lengths[tid];
    attb[tid] = att[tid];                  // att row for t = 0 into buffer 0
  }
  __syncthreads();

  const int lane  = tid & 31;
  const int wave  = tid >> 5;
  const int colin = lane & 15;
  const int rofs  = (lane & 16) ? 8 : 0;
  const int m     = wave >> 1;             // shared M-tile for this wave
  const int nbase = (wave & 1) * 4;        // 4 consecutive N-tiles
  const int arow  = 16 * m + (lane & 15);  // A-operand row for this lane

  v8f hreg[4];                // fp32 hidden state, register resident
  v8f uacc[4];                // attention-scaled update gate between phases
  for (int j = 0; j < 4; ++j)
    for (int v = 0; v < 8; ++v) hreg[j][v] = 0.0f;

  for (int t = 0; t < TT; ++t) {
    __syncthreads();  // h16/attb (written late in t-1) and rh16 reuse hazards

    // ---- Phase 1: r and u gates ------------------------------------------
    v8f ar[4], au[4];
    for (int j = 0; j < 4; ++j) {
      const size_t fb = frag_base(t, m, nbase + j, lane);
      Frag8 fr, fu;
      fr.q[0] = *(const float4*)(xr + fb); fr.q[1] = *(const float4*)(xr + fb + 4);
      fu.q[0] = *(const float4*)(xu + fb); fu.q[1] = *(const float4*)(xu + fb + 4);
      ar[j] = fr.v; au[j] = fu.v;
    }
    for (int kk = 0; kk < 4; ++kk) {
      v16h a = load_a_tile(h16, arow, 32 * kk, lane);   // one A feeds 8 WMMAs
      for (int j = 0; j < 4; ++j) {
        const int col = (nbase + j) * 16 + colin;
        ar[j] = wmma_f16(a, load_b_tile(wtr, col, 32 * kk, lane), ar[j]);
        au[j] = wmma_f16(a, load_b_tile(wtu, col, 32 * kk, lane), au[j]);
      }
    }
    const float* arow_t = attb + (t & 1) * 256;
    for (int j = 0; j < 4; ++j) {
      const int col = (nbase + j) * 16 + colin;
      for (int v = 0; v < 8; ++v) {
        const int row = 16 * m + rofs + v;
        float r = sigmoidf_fast(ar[j][v]);
        rh16[row * 128 + col] = (_Float16)(r * hreg[j][v]);
        uacc[j][v] = arow_t[row] * sigmoidf_fast(au[j][v]);
      }
    }
    __syncthreads();  // rh16 fully written before used as A operand

    // Prefetch next step's attention row into the other LDS buffer.
    if (t + 1 < TT && tid < BB)
      attb[((t + 1) & 1) * 256 + tid] = att[(size_t)(t + 1) * BB + tid];

    // ---- Phase 2: h_hat, blend, mask, output -----------------------------
    v8f ah[4];
    for (int j = 0; j < 4; ++j) {
      const size_t fb = frag_base(t, m, nbase + j, lane);
      Frag8 fh;
      fh.q[0] = *(const float4*)(xh + fb); fh.q[1] = *(const float4*)(xh + fb + 4);
      ah[j] = fh.v;
    }
    for (int kk = 0; kk < 4; ++kk) {
      v16h a = load_a_tile(rh16, arow, 32 * kk, lane);  // one A feeds 4 WMMAs
      for (int j = 0; j < 4; ++j) {
        const int col = (nbase + j) * 16 + colin;
        ah[j] = wmma_f16(a, load_b_tile(wth, col, 32 * kk, lane), ah[j]);
      }
    }
    for (int j = 0; j < 4; ++j) {
      const int col = (nbase + j) * 16 + colin;
      for (int v = 0; v < 8; ++v) {
        const int row  = 16 * m + rofs + v;
        float hhat  = tanhf(ah[j][v]);
        float u     = uacc[j][v];
        float hn    = (1.0f - u) * hreg[j][v] + u * hhat;
        bool  msk   = (t < lens[row]);
        float hkeep = msk ? hn : hreg[j][v];
        hreg[j][v] = hkeep;
        out[((size_t)t * BB + row) * HH + col] = msk ? hn : 0.0f;
        h16[row * 128 + col] = (_Float16)hkeep;   // fp16 copy for next step's A
      }
    }
  }
}

// ---------------------------------------------------------------------------
// Host launcher
// ---------------------------------------------------------------------------
extern "C" void kernel_launch(void* const* d_in, const int* in_sizes, int n_in,
                              void* d_out, int out_size, void* d_ws, size_t ws_size,
                              hipStream_t stream) {
  (void)in_sizes; (void)n_in; (void)out_size; (void)ws_size;

  const float* inputs  = (const float*)d_in[0];   // [T, B, D]
  const float* att     = (const float*)d_in[1];   // [T, B]
  const int*   lengths = (const int*)  d_in[2];   // [B]
  const float* Wr = (const float*)d_in[3];
  const float* br = (const float*)d_in[4];
  const float* Wu = (const float*)d_in[5];
  const float* bu = (const float*)d_in[6];
  const float* Wh = (const float*)d_in[7];
  const float* bh = (const float*)d_in[8];
  float* out = (float*)d_out;                     // [T, B, H]

  const size_t TBH = (size_t)TT * BB * HH;        // fragment arrays, same size
  float* xr = (float*)d_ws;       // 3 x T*B*H fp32 input-half gate pre-activations
  float* xu = xr + TBH;
  float* xh = xu + TBH;

  const int smemA = 4 * 128 * 128 * (int)sizeof(_Float16);          // 128 KB
  const int smemB = (2 * 256 * 128 + 3 * 128 * 128) * (int)sizeof(_Float16)
                    + 256 * (int)sizeof(int)                         // lens
                    + 2 * 256 * (int)sizeof(float);                  // attb

  (void)hipFuncSetAttribute(reinterpret_cast<const void*>(rnn_xpart_kernel),
                            hipFuncAttributeMaxDynamicSharedMemorySize, smemA);
  (void)hipFuncSetAttribute(reinterpret_cast<const void*>(rnn_recurrent_kernel),
                            hipFuncAttributeMaxDynamicSharedMemorySize, smemB);

  // Phase A: 1024 row-blocks, all three gates per block (x read from HBM once).
  dim3 gridA((TT * BB) / 128, 1, 1);
  rnn_xpart_kernel<<<gridA, 256, smemA, stream>>>(inputs, Wr, Wu, Wh, br, bu, bh,
                                                  xr, xu, xh);

  // Phase B: persistent single workgroup runs the 512-step recurrence.
  rnn_recurrent_kernel<<<dim3(1, 1, 1), 1024, smemB, stream>>>(
      att, lengths, xr, xu, xh, Wr, Wu, Wh, out);
}
